// RWKV6DeltaTimeMix_69818988364384
// MI455X (gfx1250) — matrix operand order repaired and checked by reading, not compiled, not run on hardware
//
#include <hip/hip_runtime.h>
#include <hip/hip_bf16.h>

// ---------------------------------------------------------------------------
// RWKV7-style chunked delta-rule time mix for gfx1250 (MI455X).
// All dense math via v_wmma_f32_16x16x32_bf16 (bf16 operands, f32 accum).
// ---------------------------------------------------------------------------

#define B_      2
#define T_      4096
#define C_      2048
#define H_      32
#define K_      64
#define CHUNK_  128
#define NC_     32
#define NTOK_   (B_ * T_)          // 8192
#define GN_EPS_ 6.4e-4f            // 1e-5 * 8^2

typedef __bf16 bf16_t;
typedef __attribute__((ext_vector_type(16))) __bf16 v16bf;
typedef __attribute__((ext_vector_type(8)))  __bf16 v8bf;
typedef __attribute__((ext_vector_type(8)))  float  v8f;

__device__ __forceinline__ float clip30(float x) {
    return fminf(30.f, fmaxf(-30.f, x));
}

// WMMA wrapper:  D = A(16x32 bf16) x B(32x16 bf16) + C(16x16 f32)
__device__ __forceinline__ v8f wmma_bf16(v16bf a, v16bf b, v8f c) {
    return __builtin_amdgcn_wmma_f32_16x16x32_bf16(false, a, false, b,
                                                   (short)0, c, false, false);
}

// A-fragment (16x32, bf16).  Lane L: M = L&15, half = L>>4.
// elements e<8 -> K = 8*half+e ; e>=8 -> K = 16+8*half+(e-8).
__device__ __forceinline__ v16bf load_a_frag(const bf16_t* p, int ld) {
    int lane = threadIdx.x & 31;
    int m = lane & 15, hh = lane >> 4;
    const bf16_t* r0 = p + (size_t)m * ld + 8 * hh;
    v8bf lo = *(const v8bf*)r0;
    v8bf hi = *(const v8bf*)(r0 + 16);
    return __builtin_shufflevector(lo, hi, 0,1,2,3,4,5,6,7,8,9,10,11,12,13,14,15);
}

// B-fragment (32x16, bf16) read from a TRANSPOSED store Bt[n][k] (row-major, ld).
// Lane L: N = L&15, half = L>>4; e<8 -> K = 8*half+e ; e>=8 -> K = 16+8*half+(e-8).
__device__ __forceinline__ v16bf load_b_frag(const bf16_t* p, int ld) {
    int lane = threadIdx.x & 31;
    int n = lane & 15, hh = lane >> 4;
    const bf16_t* r = p + (size_t)n * ld + 8 * hh;
    v8bf lo = *(const v8bf*)r;
    v8bf hi = *(const v8bf*)(r + 16);
    return __builtin_shufflevector(lo, hi, 0,1,2,3,4,5,6,7,8,9,10,11,12,13,14,15);
}

// A-fragment built from f32 source (converted on the fly).
__device__ __forceinline__ v16bf load_a_frag_f32(const float* p, int ld) {
    int lane = threadIdx.x & 31;
    int m = lane & 15, hh = lane >> 4;
    const float* r0 = p + (size_t)m * ld + 8 * hh;
    const float* r1 = r0 + 16;
    v16bf f;
#pragma unroll
    for (int j = 0; j < 8; j++) { f[j] = (bf16_t)r0[j]; f[8 + j] = (bf16_t)r1[j]; }
    return f;
}

// ---------------------------------------------------------------------------
// Generic bf16 GEMM:  C(f32, MxN) = A(bf16, MxK, lda) * B(bf16, KxN, ldb)
// Requires M % 128 == 0, K % 32 == 0.  N bounds-checked (block-uniform fast
// path).  Block = 256 threads = 8 waves; tile 128x128x32, LDS double-buffered
// (one barrier per k-step, global prefetch overlaps WMMA).
// ---------------------------------------------------------------------------
#define LDSA_ 40   // padded k-stride (bf16 elems): 80B rows -> b128-aligned, bank-spread
#define LDSB_ 40

__global__ __launch_bounds__(256)
void gemm_bf16_f32(const bf16_t* __restrict__ A, int lda,
                   const bf16_t* __restrict__ B, int ldb,
                   float* __restrict__ Cout, int ldc,
                   int M, int N, int K) {
    __shared__ alignas(16) bf16_t sA[2][128][LDSA_];
    __shared__ alignas(16) bf16_t sBt[2][128][LDSB_];

    int tid = threadIdx.x;
    int m0 = blockIdx.y * 128, n0 = blockIdx.x * 128;
    int wid = tid >> 5, lane = tid & 31;
    int wm = (wid >> 1) * 32;   // 4 waves over M
    int wn = (wid & 1) * 64;    // 2 waves over N
    bool nfull = (n0 + 128) <= N;

    int ar  = tid >> 1, acs = (tid & 1) * 16;   // A staging: row / col-seg
    int bkr = tid >> 3, bcs = (tid & 7) * 16;   // B staging: k-row / col-seg

    v8bf a0v, a1v, b0v, b1v;
    auto prefetch = [&](int k0) {
        const bf16_t* asrc = A + (size_t)(m0 + ar) * lda + (k0 + acs);
        a0v = *(const v8bf*)asrc;
        a1v = *(const v8bf*)(asrc + 8);
        const bf16_t* bsrc = B + (size_t)(k0 + bkr) * ldb + (n0 + bcs);
        if (nfull) {
            b0v = *(const v8bf*)bsrc;
            b1v = *(const v8bf*)(bsrc + 8);
        } else {
#pragma unroll
            for (int j = 0; j < 8; j++) b0v[j] = (n0 + bcs + j     < N) ? bsrc[j]     : (bf16_t)0.f;
#pragma unroll
            for (int j = 0; j < 8; j++) b1v[j] = (n0 + bcs + 8 + j < N) ? bsrc[j + 8] : (bf16_t)0.f;
        }
    };
    auto stage = [&](int buf) {
        *(v8bf*)&sA[buf][ar][acs]     = a0v;
        *(v8bf*)&sA[buf][ar][acs + 8] = a1v;
#pragma unroll
        for (int j = 0; j < 8; j++) sBt[buf][bcs + j][bkr]     = b0v[j];
#pragma unroll
        for (int j = 0; j < 8; j++) sBt[buf][bcs + 8 + j][bkr] = b1v[j];
    };

    v8f acc[2][4];
#pragma unroll
    for (int i = 0; i < 2; i++)
#pragma unroll
        for (int j = 0; j < 4; j++)
            acc[i][j] = (v8f){0.f,0.f,0.f,0.f,0.f,0.f,0.f,0.f};

    prefetch(0);
    stage(0);
    __syncthreads();
    int cur = 0;

    for (int k0 = 0; k0 < K; k0 += 32) {
        bool more = (k0 + 32) < K;
        if (more) prefetch(k0 + 32);   // global loads overlap WMMA below

        v16bf af0 = load_a_frag(&sA[cur][wm][0],      LDSA_);
        v16bf af1 = load_a_frag(&sA[cur][wm + 16][0], LDSA_);
#pragma unroll
        for (int jn = 0; jn < 4; jn++) {
            v16bf bf = load_b_frag(&sBt[cur][wn + jn * 16][0], LDSB_);
            acc[0][jn] = wmma_bf16(af0, bf, acc[0][jn]);
            acc[1][jn] = wmma_bf16(af1, bf, acc[1][jn]);
        }
        if (more) {
            stage(cur ^ 1);            // other buffer: no race with compute(cur)
            __syncthreads();
            cur ^= 1;
        }
    }

    int cn = lane & 15, hh = lane >> 4;
#pragma unroll
    for (int im = 0; im < 2; im++)
#pragma unroll
        for (int jn = 0; jn < 4; jn++)
#pragma unroll
            for (int i = 0; i < 8; i++) {
                int gm = m0 + wm + im * 16 + i + 8 * hh;
                int gn = n0 + wn + jn * 16 + cn;
                if (gn < N) Cout[(size_t)gm * ldc + gn] = acc[im][jn][i];
            }
}

// ---------------------------------------------------------------------------
// Elementwise helpers
// ---------------------------------------------------------------------------
__global__ __launch_bounds__(256)
void k_f32_to_bf16(const float* __restrict__ in, bf16_t* __restrict__ out, int n) {
    int i = blockIdx.x * 256 + threadIdx.x;
    if (i < n) out[i] = (bf16_t)in[i];
}

__global__ __launch_bounds__(256)
void k_tanh_to_bf16(const float* __restrict__ in, bf16_t* __restrict__ out, int n) {
    int i = blockIdx.x * 256 + threadIdx.x;
    if (i < n) out[i] = (bf16_t)tanhf(in[i]);
}

__global__ __launch_bounds__(256)
void k_silu_inplace(float* __restrict__ p, int n) {
    int i = blockIdx.x * 256 + threadIdx.x;
    if (i < n) { float v = p[i]; p[i] = v / (1.f + __expf(-v)); }
}

__global__ __launch_bounds__(256)
void k_neg_exp_bias(float* __restrict__ p, const float* __restrict__ td, int n) {
    int i = blockIdx.x * 256 + threadIdx.x;
    if (i < n) { int c = i % C_; p[i] = -__expf(td[c] + p[i]); }
}

__global__ __launch_bounds__(256)
void k_sigmoid_bias(float* __restrict__ p, const float* __restrict__ a0, int n) {
    int i = blockIdx.x * 256 + threadIdx.x;
    if (i < n) { int c = i % C_; p[i] = 1.f / (1.f + __expf(-(a0[c] + p[i]))); }
}

// xmix = x + (xprev - x) * time_maa_x   (bf16)
__global__ __launch_bounds__(256)
void k_xmix(const float* __restrict__ x, const float* __restrict__ shift,
            const float* __restrict__ maax, bf16_t* __restrict__ out) {
    int i = blockIdx.x * 256 + threadIdx.x;
    if (i >= NTOK_ * C_) return;
    int c = i % C_;
    int t = (i / C_) % T_;
    int b = i / (C_ * T_);
    float xv = x[i];
    float xp = t ? x[i - C_] : shift[b * C_ + c];
    out[i] = (bf16_t)(xv + (xp - xv) * maax[c]);
}

// produce the six mixed bf16 streams xw,xk,xv,xr,xg,xa
__global__ __launch_bounds__(256)
void k_mix6(const float* __restrict__ x, const float* __restrict__ shift,
            const float* __restrict__ mbase, bf16_t* __restrict__ xout,
            const float* __restrict__ mw, const float* __restrict__ mk,
            const float* __restrict__ mv, const float* __restrict__ mr,
            const float* __restrict__ mg, const float* __restrict__ ma) {
    int i = blockIdx.x * 256 + threadIdx.x;
    if (i >= NTOK_ * C_) return;
    int c = i % C_;
    int t = (i / C_) % T_;
    int b = i / (C_ * T_);
    const size_t S = (size_t)NTOK_ * C_;
    float xv = x[i];
    float xp = t ? x[i - C_] : shift[b * C_ + c];
    float dx = xp - xv;
    xout[i]         = (bf16_t)(xv + dx * (mw[c] + mbase[i]));
    xout[i + S]     = (bf16_t)(xv + dx * (mk[c] + mbase[i + S]));
    xout[i + 2*S]   = (bf16_t)(xv + dx * (mv[c] + mbase[i + 2*S]));
    xout[i + 3*S]   = (bf16_t)(xv + dx * (mr[c] + mbase[i + 3*S]));
    xout[i + 4*S]   = (bf16_t)(xv + dx * (mg[c] + mbase[i + 4*S]));
    xout[i + 5*S]   = (bf16_t)(xv + dx * (ma[c] + mbase[i + 5*S]));
}

// per (b,t,h): kk = normalize(k * k_k); k <- k * (1 + (iclr-1)*k_a)
__global__ __launch_bounds__(64)
void k_kk_norm(float* __restrict__ kbuf, const float* __restrict__ k_k,
               const float* __restrict__ k_a, const float* __restrict__ iclr,
               float* __restrict__ kkbuf) {
    int gi = blockIdx.x;              // (b*T + t)*H + h
    int h = gi % H_;
    int bt = gi / H_;
    int lane = threadIdx.x;           // 0..63
    int c = h * K_ + lane;
    size_t idx = (size_t)bt * C_ + c;
    float kv_ = kbuf[idx];
    float tmp = kv_ * k_k[c];
    __shared__ float red[64];
    red[lane] = tmp * tmp;
    __syncthreads();
    for (int off = 32; off > 0; off >>= 1) {
        if (lane < off) red[lane] += red[lane + off];
        __syncthreads();
    }
    float nrm = fmaxf(sqrtf(red[0]), 1e-12f);
    float ic = iclr[idx];
    kkbuf[idx] = tmp / nrm;
    kbuf[idx]  = kv_ * (1.f + (ic - 1.f) * k_a[c]);
}

// shift_state_out = x[:, T-1, :]
__global__ __launch_bounds__(256)
void k_shift_out(const float* __restrict__ x, float* __restrict__ out) {
    int i = blockIdx.x * 256 + threadIdx.x;
    if (i >= B_ * C_) return;
    int b = i / C_, c = i % C_;
    out[i] = x[((size_t)(b * T_ + T_ - 1)) * C_ + c];
}

// ---------------------------------------------------------------------------
// WKV chunk kernel (parallel over B*NC*H = 2048 chunk-heads).
// LDS strides: LDQ_=72 for [128][64] operand tiles, LDT_=136 for [64][128]
// transposed tiles and the 128x128 A/corr tiles (b128-aligned, bank-spread).
// ---------------------------------------------------------------------------
#define LDQ_ 72
#define LDT_ 136

__global__ __launch_bounds__(256)
void wkv_chunk_kernel(const float* __restrict__ r_g, const float* __restrict__ k_g,
                      const float* __restrict__ v_g, const float* __restrict__ w_g,
                      const float* __restrict__ kkg, const float* __restrict__ icg,
                      const float* __restrict__ u_g,
                      float* __restrict__ y_g, bf16_t* __restrict__ rw_g,
                      float* __restrict__ kv_g, bf16_t* __restrict__ abT_g,
                      float* __restrict__ ws_g) {
    int bid = blockIdx.x;
    int h = bid % H_;
    int n = (bid / H_) % NC_;
    int b = bid / (H_ * NC_);
    size_t base = ((size_t)(b * T_ + n * CHUNK_)) * C_ + h * K_;  // row stride C_

    __shared__ alignas(16) float   s_wcum[128][64];                 // 32 KB
    __shared__ alignas(16) bf16_t  s_A[128][LDT_];                  // 34 KB
    __shared__ alignas(16) unsigned char s_u1[128 * LDT_ * 2];      // 34 KB union
    __shared__ alignas(16) bf16_t  bufA[128 * LDQ_];                // 18 KB (>= 64*LDT_)
    __shared__ alignas(16) bf16_t  bufB[128 * LDQ_];                // 18 KB
    __shared__ float s_off[64], s_wsum[64], s_diag[128], s_dp[128][2];

    float* s_w = (float*)s_u1;                        // phases 1-2 ([128][64] f32)
    bf16_t (*s_corrT)[LDT_] = (bf16_t(*)[LDT_])s_u1;  // phases 5+

    int tid = threadIdx.x, wid = tid >> 5, lane = tid & 31;
    const v8f vz = (v8f){0.f,0.f,0.f,0.f,0.f,0.f,0.f,0.f};

    // ---- P1: load w_log tile, column-wise cumsum ----
    for (int i = tid; i < 128 * 64; i += 256) {
        int t = i >> 6, kx = i & 63;
        s_w[i] = w_g[base + (size_t)t * C_ + kx];
    }
    __syncthreads();
    if (tid < 64) {
        float cum = 0.f;
        for (int t = 0; t < 128; t++) { cum += s_w[t * 64 + tid]; s_wcum[t][tid] = cum; }
        s_off[tid]  = s_wcum[63][tid];        // exclusive cumsum at mid
        s_wsum[tid] = cum;
        ws_g[(size_t)bid * 64 + tid] = __expf(cum);
    }
    __syncthreads();

    // ---- P2: qd = r * exp(clip(wcs - offset)) ; rw = r * exp(clip(wcs)) ----
    for (int i = tid; i < 128 * 64; i += 256) {
        int t = i >> 6, kx = i & 63;
        float rv  = r_g[base + (size_t)t * C_ + kx];
        float wcs = s_wcum[t][kx] - s_w[i];
        bufA[t * LDQ_ + kx] = (bf16_t)(rv * __expf(clip30(wcs - s_off[kx])));
        rw_g[base + (size_t)t * C_ + kx] = (bf16_t)(rv * __expf(clip30(wcs)));
    }
    // ---- P3: kid = k * exp(clip(offset - wcum)) ; diag partials ----
    {
        int t = tid >> 1;
        int kstart = (tid & 1) * 32;
        const float* u_h = u_g + h * 64;
        float part = 0.f;
        for (int j = 0; j < 32; j++) {
            int kx = kstart + j;
            float kv_  = k_g[base + (size_t)t * C_ + kx];
            float kinv = __expf(clip30(s_off[kx] - s_wcum[t][kx]));
            bufB[t * LDQ_ + kx] = (bf16_t)(kv_ * kinv);
            part += r_g[base + (size_t)t * C_ + kx] * u_h[kx] * kv_;
        }
        s_dp[t][tid & 1] = part;
    }
    __syncthreads();
    if (tid < 128) s_diag[tid] = s_dp[tid][0] + s_dp[tid][1];
    __syncthreads();

    // ---- P4: A = tril(qd @ kid^T, -1) + diag*I  -> s_A (bf16) ----
    {
        int t0 = wid * 16;
        int cn = lane & 15, hh = lane >> 4;
        for (int jt = 0; jt < 8; jt++) {
            v8f acc = vz;
#pragma unroll
            for (int k0 = 0; k0 < 64; k0 += 32) {
                v16bf af = load_a_frag(&bufA[t0 * LDQ_ + k0], LDQ_);
                v16bf bf = load_b_frag(&bufB[jt * 16 * LDQ_ + k0], LDQ_);
                acc = wmma_bf16(af, bf, acc);
            }
#pragma unroll
            for (int i = 0; i < 8; i++) {
                int m  = t0 + i + 8 * hh;
                int nn = jt * 16 + cn;
                float val = (m > nn) ? acc[i] : 0.f;
                if (m == nn) val = s_diag[m];
                s_A[m][nn] = (bf16_t)val;
            }
        }
    }
    __syncthreads();

    // ---- P5: kki (bufA), kk (bufB); corrT = tril(kki@kk^T,-1) transposed ----
    for (int i = tid; i < 128 * 64; i += 256) {
        int t = i >> 6, kx = i & 63;
        float kkv = kkg[base + (size_t)t * C_ + kx];
        float ic  = icg[base + (size_t)t * C_ + kx];
        bufA[t * LDQ_ + kx] = (bf16_t)(kkv * ic);
        bufB[t * LDQ_ + kx] = (bf16_t)kkv;
    }
    __syncthreads();
    {
        int s0 = wid * 16;
        int cn = lane & 15, hh = lane >> 4;
        for (int jt = 0; jt < 8; jt++) {
            v8f acc = vz;
#pragma unroll
            for (int k0 = 0; k0 < 64; k0 += 32) {
                v16bf af = load_a_frag(&bufA[s0 * LDQ_ + k0], LDQ_);
                v16bf bf = load_b_frag(&bufB[jt * 16 * LDQ_ + k0], LDQ_);
                acc = wmma_bf16(af, bf, acc);
            }
#pragma unroll
            for (int i = 0; i < 8; i++) {
                int ss = s0 + i + 8 * hh;
                int jj = jt * 16 + cn;
                s_corrT[jj][ss] = (bf16_t)((ss > jj) ? acc[i] : 0.f);
            }
        }
    }
    __syncthreads();

    // ---- P6: A_total = A - tril(A @ corr) (in place, rows owned per wave) ----
    {
        int t0 = wid * 16;
        int cn = lane & 15, hh = lane >> 4;
        v8f Tacc[8];
#pragma unroll
        for (int jt = 0; jt < 8; jt++) Tacc[jt] = vz;
        for (int k0 = 0; k0 < 128; k0 += 32) {
            v16bf af = load_a_frag(&s_A[t0][0] + k0, LDT_);
#pragma unroll
            for (int jt = 0; jt < 8; jt++) {
                v16bf bf = load_b_frag(&s_corrT[jt * 16][0] + k0, LDT_);
                Tacc[jt] = wmma_bf16(af, bf, Tacc[jt]);
            }
        }
#pragma unroll
        for (int jt = 0; jt < 8; jt++)
#pragma unroll
            for (int i = 0; i < 8; i++) {
                int m = t0 + i + 8 * hh, nn = jt * 16 + cn;
                float a = (float)s_A[m][nn];
                s_A[m][nn] = (bf16_t)(a - ((m >= nn) ? Tacc[jt][i] : 0.f));
            }
    }
    __syncthreads();

    // ---- P7: vT (bufA as [64][LDT_]); y_intra = A_total @ v ----
    bf16_t* vT = bufA;
    for (int i = tid; i < 128 * 64; i += 256) {
        int t = i >> 6, kx = i & 63;
        vT[kx * LDT_ + t] = (bf16_t)v_g[base + (size_t)t * C_ + kx];
    }
    __syncthreads();
    {
        int t0 = wid * 16;
        int cn = lane & 15, hh = lane >> 4;
        for (int ct = 0; ct < 4; ct++) {
            v8f acc = vz;
#pragma unroll
            for (int k0 = 0; k0 < 128; k0 += 32) {
                v16bf af = load_a_frag(&s_A[t0][0] + k0, LDT_);
                v16bf bf = load_b_frag(&vT[ct * 16 * LDT_ + k0], LDT_);
                acc = wmma_bf16(af, bf, acc);
            }
#pragma unroll
            for (int i = 0; i < 8; i++) {
                int m = t0 + i + 8 * hh;
                y_g[base + (size_t)m * C_ + ct * 16 + cn] = acc[i];
            }
        }
    }
    __syncthreads();

    // ---- P8: kwiT (bufB as [64][LDT_]); kv = kwi^T @ v ----
    bf16_t* kwiT = bufB;
    for (int i = tid; i < 128 * 64; i += 256) {
        int t = i >> 6, kx = i & 63;
        float kv_  = k_g[base + (size_t)t * C_ + kx];
        float wint = __expf(clip30(s_wsum[kx] - s_wcum[t][kx]));
        kwiT[kx * LDT_ + t] = (bf16_t)(kv_ * wint);
    }
    __syncthreads();
    {
        int cn = lane & 15, hh = lane >> 4;
        for (int q = 0; q < 2; q++) {
            int tile = wid * 2 + q, mi = tile >> 2, ni = tile & 3;
            v8f acc = vz;
#pragma unroll
            for (int k0 = 0; k0 < 128; k0 += 32) {
                v16bf af = load_a_frag(&kwiT[mi * 16 * LDT_ + k0], LDT_);
                v16bf bf = load_b_frag(&vT[ni * 16 * LDT_ + k0], LDT_);
                acc = wmma_bf16(af, bf, acc);
            }
#pragma unroll
            for (int i = 0; i < 8; i++) {
                int kd = mi * 16 + i + 8 * hh, vd = ni * 16 + cn;
                kv_g[(size_t)bid * 4096 + kd * 64 + vd] = acc[i];
            }
        }
    }
    __syncthreads();

    // ---- P9: nkwiT (bufA), kkiT (bufB); ab = (-kk*w_inter)^T @ (kk*iclr) ----
    bf16_t* nkwiT = bufA;
    bf16_t* kkiT  = bufB;
    for (int i = tid; i < 128 * 64; i += 256) {
        int t = i >> 6, kx = i & 63;
        float kkv  = kkg[base + (size_t)t * C_ + kx];
        float ic   = icg[base + (size_t)t * C_ + kx];
        float wint = __expf(clip30(s_wsum[kx] - s_wcum[t][kx]));
        nkwiT[kx * LDT_ + t] = (bf16_t)(-kkv * wint);
        kkiT[kx * LDT_ + t]  = (bf16_t)(kkv * ic);
    }
    __syncthreads();
    {
        int cn = lane & 15, hh = lane >> 4;
        for (int q = 0; q < 2; q++) {
            int tile = wid * 2 + q, mi = tile >> 2, ni = tile & 3;
            v8f acc = vz;
#pragma unroll
            for (int k0 = 0; k0 < 128; k0 += 32) {
                v16bf af = load_a_frag(&nkwiT[mi * 16 * LDT_ + k0], LDT_);
                v16bf bf = load_b_frag(&kkiT[ni * 16 * LDT_ + k0], LDT_);
                acc = wmma_bf16(af, bf, acc);
            }
#pragma unroll
            for (int i = 0; i < 8; i++) {
                int kd = mi * 16 + i + 8 * hh, jd = ni * 16 + cn;
                abT_g[(size_t)bid * 4096 + jd * 64 + kd] = (bf16_t)acc[i];  // transposed
            }
        }
    }
}

// ---------------------------------------------------------------------------
// WKV state kernel (sequential over 32 chunks; parallel over B*H = 64 blocks).
// ---------------------------------------------------------------------------
#define LDST_ 68   // f32 state row stride (272B: b128-aligned, bank-spread)
#define LDSTT_ 72  // bf16 transposed-state row stride (144B)

__global__ __launch_bounds__(256)
void wkv_state_kernel(const float* __restrict__ state_in,
                      const float* __restrict__ y_in,
                      const bf16_t* __restrict__ rw_g,
                      const float* __restrict__ kv_g,
                      const bf16_t* __restrict__ abT_g,
                      const float* __restrict__ ws_g,
                      const float* __restrict__ g_g,
                      const float* __restrict__ gnw,
                      const float* __restrict__ gnb,
                      bf16_t* __restrict__ ygn_g,
                      float* __restrict__ state_out) {
    int blk = blockIdx.x;
    int h = blk % H_, b = blk / H_;
    __shared__ alignas(16) float  st[2][64][LDST_];   // ping-pong f32 state
    __shared__ alignas(16) bf16_t stT[64][LDSTT_];    // transposed bf16 state
    __shared__ alignas(16) float  ys[8][16][64];      // per-wave y strips
    __shared__ float mu[8][16], rs[8][16];

    int tid = threadIdx.x, wid = tid >> 5, lane = tid & 31;
    const v8f vz = (v8f){0.f,0.f,0.f,0.f,0.f,0.f,0.f,0.f};
    int cur = 0;

    for (int i = tid; i < 4096; i += 256)
        st[0][i >> 6][i & 63] = state_in[((size_t)(b * H_ + h)) * 4096 + i];
    __syncthreads();

    for (int c = 0; c < NC_; c++) {
        size_t idx = ((size_t)(b * NC_ + c)) * H_ + h;
        // transpose + convert current state for B-operand use
        for (int i = tid; i < 4096; i += 256) {
            int kx = i >> 6, vx = i & 63;
            stT[vx][kx] = (bf16_t)st[cur][kx][vx];
        }
        __syncthreads();

        // ---- y = y_intra + rw @ state ; fused GroupNorm * g ----
        {
            int t0 = wid * 16;
            int cn = lane & 15, hh = lane >> 4;
            size_t rowbase = ((size_t)(b * T_ + c * CHUNK_ + t0)) * C_ + h * K_;
            for (int ct = 0; ct < 4; ct++) {
                v8f acc = vz;
#pragma unroll
                for (int k0 = 0; k0 < 64; k0 += 32) {
                    v16bf af = load_a_frag(rw_g + rowbase + k0, C_);
                    v16bf bf = load_b_frag(&stT[ct * 16][0] + k0, LDSTT_);
                    acc = wmma_bf16(af, bf, acc);
                }
#pragma unroll
                for (int i = 0; i < 8; i++) {
                    int rr = i + 8 * hh;
                    float yv = acc[i] + y_in[rowbase + (size_t)rr * C_ + ct * 16 + cn];
                    ys[wid][rr][ct * 16 + cn] = yv;
                }
            }
            // per-row stats (wave-local; LDS ops are in-order within a wave)
            if (lane < 16) {
                float s1 = 0.f, s2 = 0.f;
                for (int j = 0; j < 64; j++) {
                    float v = ys[wid][lane][j];
                    s1 += v; s2 += v * v;
                }
                float m_ = s1 * (1.f / 64.f);
                float var = s2 * (1.f / 64.f) - m_ * m_;
                mu[wid][lane] = m_;
                rs[wid][lane] = rsqrtf(var + GN_EPS_);
            }
            for (int i = lane; i < 16 * 64; i += 32) {
                int rr = i >> 6, cc = i & 63;
                float v = (ys[wid][rr][cc] - mu[wid][rr]) * rs[wid][rr];
                int cg = h * K_ + cc;
                v = v * gnw[cg] + gnb[cg];
                size_t gi = ((size_t)(b * T_ + c * CHUNK_ + t0 + rr)) * C_ + cg;
                v *= g_g[gi];
                ygn_g[gi] = (bf16_t)v;
            }
        }
        __syncthreads();

        // ---- state = state*ws + state@ab + kv ----
        {
            int cn = lane & 15, hh = lane >> 4;
            for (int q = 0; q < 2; q++) {
                int tile = wid * 2 + q, mi = tile >> 2, ni = tile & 3;
                v8f acc = vz;
#pragma unroll
                for (int k0 = 0; k0 < 64; k0 += 32) {
                    v16bf af = load_a_frag_f32(&st[cur][mi * 16][0] + k0, LDST_);
                    v16bf bf = load_b_frag(abT_g + idx * 4096 + (size_t)(ni * 16) * 64 + k0, 64);
                    acc = wmma_bf16(af, bf, acc);
                }
#pragma unroll
                for (int i = 0; i < 8; i++) {
                    int kd = mi * 16 + i + 8 * hh, jd = ni * 16 + cn;
                    float old = st[cur][kd][jd];
                    float wsj = ws_g[idx * 64 + jd];
                    st[cur ^ 1][kd][jd] =
                        old * wsj + acc[i] + kv_g[idx * 4096 + (size_t)kd * 64 + jd];
                }
            }
        }
        __syncthreads();
        cur ^= 1;
    }

    for (int i = tid; i < 4096; i += 256)
        state_out[((size_t)(b * H_ + h)) * 4096 + i] = st[cur][i >> 6][i & 63];
}

// ---------------------------------------------------------------------------
// Host-side launch
// ---------------------------------------------------------------------------
extern "C" void kernel_launch(void* const* d_in, const int* in_sizes, int n_in,
                              void* d_out, int out_size, void* d_ws, size_t ws_size,
                              hipStream_t stream) {
    const float* x       = (const float*)d_in[0];
    const float* shift_i = (const float*)d_in[1];
    const float* state_i = (const float*)d_in[2];
    const float* maa_x   = (const float*)d_in[3];
    const float* maa_w   = (const float*)d_in[4];
    const float* maa_k   = (const float*)d_in[5];
    const float* maa_v   = (const float*)d_in[6];
    const float* maa_r   = (const float*)d_in[7];
    const float* maa_g   = (const float*)d_in[8];
    const float* maa_a   = (const float*)d_in[9];
    const float* maa_w1  = (const float*)d_in[10];
    const float* maa_w2  = (const float*)d_in[11];
    const float* t_decay = (const float*)d_in[12];
    const float* dw1     = (const float*)d_in[13];
    const float* dw2     = (const float*)d_in[14];
    const float* faaaa   = (const float*)d_in[15];
    const float* a0      = (const float*)d_in[16];
    const float* a1      = (const float*)d_in[17];
    const float* a2      = (const float*)d_in[18];
    const float* k_k     = (const float*)d_in[19];
    const float* k_a     = (const float*)d_in[20];
    const float* W_r     = (const float*)d_in[21];
    const float* W_k     = (const float*)d_in[22];
    const float* W_v     = (const float*)d_in[23];
    const float* W_g     = (const float*)d_in[24];
    const float* W_o     = (const float*)d_in[25];
    const float* gnw     = (const float*)d_in[26];
    const float* gnb     = (const float*)d_in[27];

    char* ws = (char*)d_ws;
    size_t o = 0;
    auto alloc = [&](size_t bytes) { size_t r = o; o += (bytes + 255) & ~(size_t)255; return r; };

    const size_t TOKC  = (size_t)NTOK_ * C_;       // 16,777,216 elems
    const size_t SLOTF = TOKC * 4;                 // f32 [NTOK,C]
    const size_t SLOTB = TOKC * 2;                 // bf16 [NTOK,C]

    // bf16 weights
    size_t o_wr = alloc((size_t)C_ * C_ * 2), o_wk = alloc((size_t)C_ * C_ * 2);
    size_t o_wv = alloc((size_t)C_ * C_ * 2), o_wg = alloc((size_t)C_ * C_ * 2);
    size_t o_wo = alloc((size_t)C_ * C_ * 2);
    size_t o_w1 = alloc((size_t)C_ * 192 * 2);
    size_t o_w2 = alloc((size_t)6 * 32 * C_ * 2);
    size_t o_d1 = alloc((size_t)C_ * 64 * 2);
    size_t o_d2 = alloc((size_t)64 * C_ * 2);
    size_t o_a1 = alloc((size_t)C_ * 96 * 2);
    size_t o_a2 = alloc((size_t)96 * C_ * 2);
    // activations
    size_t o_xmix = alloc(SLOTB);                         // later reused as rw (bf16)
    size_t o_xxx  = alloc((size_t)NTOK_ * 192 * 4);
    size_t o_xxxb = alloc((size_t)NTOK_ * 192 * 2);
    size_t o_m    = alloc(6 * SLOTF);   // m[0..5]; reused: r,k,v,g,wlog,kk
    size_t o_x6   = alloc(6 * SLOTB);   // xw..xa; reused: y_intra,kv,ygn,abT,ws
    size_t o_iclr = alloc(SLOTF);
    size_t o_hd   = alloc((size_t)NTOK_ * 64 * 4);
    size_t o_hdb  = alloc((size_t)NTOK_ * 64 * 2);
    size_t o_ha   = alloc((size_t)NTOK_ * 96 * 4);
    size_t o_hab  = alloc((size_t)NTOK_ * 96 * 2);
    (void)ws_size; (void)n_in; (void)in_sizes; (void)out_size;   // needs ~770 MB ws

    bf16_t* wr_bf = (bf16_t*)(ws + o_wr); bf16_t* wk_bf = (bf16_t*)(ws + o_wk);
    bf16_t* wv_bf = (bf16_t*)(ws + o_wv); bf16_t* wg_bf = (bf16_t*)(ws + o_wg);
    bf16_t* wo_bf = (bf16_t*)(ws + o_wo);
    bf16_t* w1_bf = (bf16_t*)(ws + o_w1); bf16_t* w2_bf = (bf16_t*)(ws + o_w2);
    bf16_t* d1_bf = (bf16_t*)(ws + o_d1); bf16_t* d2_bf = (bf16_t*)(ws + o_d2);
    bf16_t* a1_bf = (bf16_t*)(ws + o_a1); bf16_t* a2_bf = (bf16_t*)(ws + o_a2);
    bf16_t* xmix  = (bf16_t*)(ws + o_xmix);
    bf16_t* rw_bf = xmix;                                  // alias (xmix dead)
    float*  xxx   = (float*) (ws + o_xxx);
    bf16_t* xxxb  = (bf16_t*)(ws + o_xxxb);
    float*  mbase = (float*) (ws + o_m);
    float*  rbuf  = mbase;                                 // aliases after k_mix6
    float*  kbuf  = (float*)(ws + o_m + 1 * SLOTF);
    float*  vbuf  = (float*)(ws + o_m + 2 * SLOTF);
    float*  gbuf  = (float*)(ws + o_m + 3 * SLOTF);
    float*  wlog  = (float*)(ws + o_m + 4 * SLOTF);
    float*  kkbuf = (float*)(ws + o_m + 5 * SLOTF);
    bf16_t* x6    = (bf16_t*)(ws + o_x6);
    bf16_t* xw_bf = x6;                     bf16_t* xk_bf = (bf16_t*)(ws + o_x6 + 1*SLOTB);
    bf16_t* xv_bf = (bf16_t*)(ws + o_x6 + 2*SLOTB); bf16_t* xr_bf = (bf16_t*)(ws + o_x6 + 3*SLOTB);
    bf16_t* xg_bf = (bf16_t*)(ws + o_x6 + 4*SLOTB); bf16_t* xa_bf = (bf16_t*)(ws + o_x6 + 5*SLOTB);
    // x6 region reuse after projection GEMMs:
    float*  y_intra = (float*)(ws + o_x6);                     // slots 0-1
    float*  kvbuf   = (float*)(ws + o_x6 + 2 * SLOTB);         // slot 2
    bf16_t* ygn     = (bf16_t*)(ws + o_x6 + 3 * SLOTB);        // slot 3
    bf16_t* abT     = (bf16_t*)(ws + o_x6 + 4 * SLOTB);        // slot 4 (16.8MB)
    float*  wsbuf   = (float*)(ws + o_x6 + 4 * SLOTB + (size_t)2048 * 4096 * 2);
    float*  iclr    = (float*)(ws + o_iclr);
    float*  hd      = (float*)(ws + o_hd);  bf16_t* hdb = (bf16_t*)(ws + o_hdb);
    float*  ha      = (float*)(ws + o_ha);  bf16_t* hab = (bf16_t*)(ws + o_hab);

    float* out_main  = (float*)d_out;
    float* out_shift = out_main + TOKC;
    float* out_state = out_shift + (size_t)B_ * C_;

    auto cvt = [&](const float* src, bf16_t* dst, size_t n) {
        k_f32_to_bf16<<<(int)((n + 255) / 256), 256, 0, stream>>>(src, dst, (int)n);
    };
    auto gemm = [&](const bf16_t* A, int lda, const bf16_t* B, int ldb,
                    float* Cp, int ldc, int M, int N, int K) {
        dim3 g((N + 127) / 128, (M + 127) / 128);
        gemm_bf16_f32<<<g, 256, 0, stream>>>(A, lda, B, ldb, Cp, ldc, M, N, K);
    };

    // 1) weights -> bf16
    cvt(W_r, wr_bf, (size_t)C_ * C_); cvt(W_k, wk_bf, (size_t)C_ * C_);
    cvt(W_v, wv_bf, (size_t)C_ * C_); cvt(W_g, wg_bf, (size_t)C_ * C_);
    cvt(W_o, wo_bf, (size_t)C_ * C_);
    cvt(maa_w1, w1_bf, (size_t)C_ * 192); cvt(maa_w2, w2_bf, (size_t)6 * 32 * C_);
    cvt(dw1, d1_bf, (size_t)C_ * 64);     cvt(dw2, d2_bf, (size_t)64 * C_);
    cvt(a1, a1_bf, (size_t)C_ * 96);      cvt(a2, a2_bf, (size_t)96 * C_);

    // 2) token-shift mix input + w1 GEMM + tanh
    k_xmix<<<(int)(TOKC / 256), 256, 0, stream>>>(x, shift_i, maa_x, xmix);
    gemm(xmix, C_, w1_bf, 192, xxx, 192, NTOK_, 192, C_);
    k_tanh_to_bf16<<<(int)((NTOK_ * 192 + 255) / 256), 256, 0, stream>>>(xxx, xxxb, NTOK_ * 192);

    // 3) six m_s = xxx_s @ w2_s
    for (int s = 0; s < 6; s++)
        gemm(xxxb + s * 32, 192, w2_bf + (size_t)s * 32 * C_, C_,
             mbase + (size_t)s * TOKC, C_, NTOK_, C_, 32);

    // 4) build xw..xa (bf16)
    k_mix6<<<(int)(TOKC / 256), 256, 0, stream>>>(x, shift_i, mbase, x6,
                                                  maa_w, maa_k, maa_v, maa_r, maa_g, maa_a);

    // 5) projections (m region reused as outputs)
    gemm(xr_bf, C_, wr_bf, C_, rbuf, C_, NTOK_, C_, C_);
    gemm(xk_bf, C_, wk_bf, C_, kbuf, C_, NTOK_, C_, C_);
    gemm(xv_bf, C_, wv_bf, C_, vbuf, C_, NTOK_, C_, C_);
    gemm(xg_bf, C_, wg_bf, C_, gbuf, C_, NTOK_, C_, C_);
    gemm(xw_bf, C_, d1_bf, 64, hd, 64, NTOK_, 64, C_);
    gemm(xa_bf, C_, a1_bf, 96, ha, 96, NTOK_, 96, C_);

    // 6) decay path: wlog = -exp(time_decay + tanh(hd)@dw2)
    k_tanh_to_bf16<<<(int)((NTOK_ * 64 + 255) / 256), 256, 0, stream>>>(hd, hdb, NTOK_ * 64);
    gemm(hdb, 64, d2_bf, C_, wlog, C_, NTOK_, C_, 64);
    k_neg_exp_bias<<<(int)(TOKC / 256), 256, 0, stream>>>(wlog, t_decay, (int)TOKC);

    // 7) iclr = sigmoid(a0 + ha@a2)
    k_f32_to_bf16<<<(int)((NTOK_ * 96 + 255) / 256), 256, 0, stream>>>(ha, hab, NTOK_ * 96);
    gemm(hab, 96, a2_bf, C_, iclr, C_, NTOK_, C_, 96);
    k_sigmoid_bias<<<(int)(TOKC / 256), 256, 0, stream>>>(iclr, a0, (int)TOKC);

    // 8) g = silu(g); kk = normalize(k*k_k); k <- k*(1+(iclr-1)*k_a)
    k_silu_inplace<<<(int)(TOKC / 256), 256, 0, stream>>>(gbuf, (int)TOKC);
    k_kk_norm<<<NTOK_ * H_, 64, 0, stream>>>(kbuf, k_k, k_a, iclr, kkbuf);

    // 9) WKV chunked delta
    wkv_chunk_kernel<<<B_ * NC_ * H_, 256, 0, stream>>>(
        rbuf, kbuf, vbuf, wlog, kkbuf, iclr, faaaa,
        y_intra, rw_bf, kvbuf, abT, wsbuf);
    wkv_state_kernel<<<B_ * H_, 256, 0, stream>>>(
        state_i, y_intra, rw_bf, kvbuf, abT, wsbuf,
        gbuf, gnw, gnb, ygn, out_state);

    // 10) out = ygn @ W_o ; shift_state_out
    gemm(ygn, C_, wo_bf, C_, out_main, C_, NTOK_, C_, C_);
    k_shift_out<<<(B_ * C_ + 255) / 256, 256, 0, stream>>>(x, out_shift);
}
